// CustomRNN_30176440222073
// MI455X (gfx1250) — matrix operand order repaired and checked by reading, not compiled, not run on hardware
//
#include <hip/hip_runtime.h>
#include <hip/hip_bf16.h>

typedef __attribute__((ext_vector_type(16))) __bf16 v16bf;
typedef __attribute__((ext_vector_type(8)))  __bf16 v8bf;
typedef __attribute__((ext_vector_type(8)))  float  v8f;

#define BB   64      // batch
#define TT   256     // seq len
#define MM   1024    // hidden
#define LL   2       // layers
#define K2   2048    // concat K = M (x part) + M (h part)
#define G4   4096    // 4*M gate rows
#define NBLK 32      // persistent grid
#define NTHR 256     // 8 waves
#define KSL  128     // K-slice staged in LDS (16 KB per buffer, double buffered)
#define NSL  (K2 / KSL)          // 16 slices
#define SLICE_BYTES ((KSL / 8) * BB * 8 * 2)   // 16384
#define SLICE_VECS  ((KSL / 8) * BB)           // 1024 x b128

// staging layout: [l][k>>3][b][k&7]  (groups of 8 K-values contiguous per column
// so a B-fragment half (8 consecutive K for one column) is one b128 load)
__device__ __forceinline__ size_t stg(int l, int k, int b) {
    return ((((size_t)l * (K2 / 8) + (k >> 3)) * BB + b) << 3) + (k & 7);
}

// CDNA5 async global->LDS DMA (ASYNCcnt), bypasses VGPRs/LOADcnt
__device__ __forceinline__ void async_ld_b128(unsigned lds_off, const void* g) {
    asm volatile("global_load_async_to_lds_b128 %0, %1, off"
                 :: "v"(lds_off), "v"(g)
                 : "memory");
}
__device__ __forceinline__ void wait_async() {
    asm volatile("s_wait_asynccnt 0x0" ::: "memory");
}

__global__ void init_bar(unsigned* bar) { *bar = 0u; }

// Wcat[l][row][k] = (k<1024) ? Wx[l][row][k] : Wh[l][row][k-1024], as bf16
__global__ void convert_weights(const float* __restrict__ Wx,
                                const float* __restrict__ Wh,
                                __bf16* __restrict__ Wcat) {
    const size_t total = (size_t)LL * G4 * K2;   // 2^24
    for (size_t i = (size_t)blockIdx.x * blockDim.x + threadIdx.x; i < total;
         i += (size_t)gridDim.x * blockDim.x) {
        const size_t l   = i >> 23;              // G4*K2 = 2^23
        const size_t rem = i & ((1u << 23) - 1);
        const size_t row = rem >> 11;            // /K2
        const size_t k   = rem & (K2 - 1);
        const float v = (k < MM) ? Wx[(l << 22) + (row << 10) + k]
                                 : Wh[(l << 22) + (row << 10) + (k - MM)];
        Wcat[i] = (__bf16)v;
    }
}

__device__ __forceinline__ void grid_sync(unsigned* bar, unsigned& epoch) {
    __threadfence();
    __syncthreads();
    ++epoch;
    if (threadIdx.x == 0) {
        atomicAdd(bar, 1u);
        const unsigned target = epoch * NBLK;
        while (*((volatile unsigned*)bar) < target)
            __builtin_amdgcn_s_sleep(2);
    }
    __syncthreads();
    __threadfence();
}

__device__ __forceinline__ float sigm(float x) {
    return 1.0f / (1.0f + __expf(-x));
}
__device__ __forceinline__ float tanh_fast(float x) {
    const float e = __expf(2.0f * x);
    return (e - 1.0f) / (e + 1.0f);
}

__global__ __launch_bounds__(NTHR) void lstm_persistent(
    const float* __restrict__ x,      // (B,T,N)
    const float* __restrict__ h0,     // (L,B,M)
    const float* __restrict__ c0,     // (L,B,M)
    const float* __restrict__ bias,   // (L,4M)
    const __bf16* __restrict__ Wcat,  // [L][4096][2048] bf16
    __bf16* __restrict__ stage,       // [L][K2/8][B][8] bf16
    float* __restrict__ c_ws,         // (L,B,M) fp32 running cell state
    unsigned* __restrict__ bar,
    float* __restrict__ out)          // outs (B,T,M) | h_n (L,B,M) | c_n (L,B,M)
{
    const int tid  = threadIdx.x;
    const int bid  = blockIdx.x;          // 0..31
    const int lane = tid & 31;
    const int wave = tid >> 5;            // 0..7
    const int hi   = lane >> 4;           // 0/1 half-wave
    const int colb = lane & 15;
    unsigned epoch = 0;

    const size_t OUTS = (size_t)BB * TT * MM;
    const size_t LBM  = (size_t)LL * BB * MM;

    __shared__ float  gl[4][32][BB];             // [gate][local m][batch] 32 KB
    __shared__ __bf16 panel[2][KSL / 8][BB][8];  // double-buffered B slice 32 KB

    // LDS byte offset of panel (flat->LDS mapping uses addr[31:0])
    const unsigned panel_base = (unsigned)(uintptr_t)&panel[0][0][0][0];

    // ---- seed: c state + bf16 h into both layers' h staging regions ----
    for (int i = bid * NTHR + tid; i < LL * BB * MM; i += NBLK * NTHR) {
        const int l = i >> 16;            // /(B*M)
        const int b = (i >> 10) & (BB - 1);
        const int m = i & (MM - 1);
        c_ws[i] = c0[i];
        stage[stg(l, MM + m, b)] = (__bf16)h0[i];
    }
    grid_sync(bar, epoch);

    for (int t = 0; t < TT; ++t) {
        // ---- phase A: stage x_t (fp32 -> bf16, transposed group-of-8) ----
        for (int i = bid * NTHR + tid; i < BB * MM; i += NBLK * NTHR) {
            const int b = i >> 10;
            const int n = i & (MM - 1);
            stage[stg(0, n, b)] = (__bf16)x[((size_t)b * TT + t) * MM + n];
        }
        grid_sync(bar, epoch);

        for (int l = 0; l < LL; ++l) {
            // ---- phase B: WMMA GEMM  gates = [x;h] @ [Wx;Wh]^T ----
            // wave -> gate section s (i/f/g/o): 32 rows x 32 batch cols
            const int s  = wave >> 1;          // gate section
            const int ch = wave & 1;           // column half (32 cols)
            const __bf16* wrow0 = Wcat +
                ((size_t)l * G4 + s * MM + bid * 32 + 0 * 16 + colb) * K2;
            const __bf16* wrow1 = Wcat +
                ((size_t)l * G4 + s * MM + bid * 32 + 1 * 16 + colb) * K2;
            const __bf16* stageL = stage + (size_t)l * (K2 / 8) * BB * 8;

            v8f acc00 = {}, acc01 = {}, acc10 = {}, acc11 = {};

            // issue async DMA of slice 0 into buffer 0
            {
                const v8bf* src = (const v8bf*)stageL;
#pragma unroll
                for (int i = 0; i < SLICE_VECS / NTHR; ++i) {
                    const unsigned vi = tid + i * NTHR;
                    async_ld_b128(panel_base + vi * 16, src + vi);
                }
            }
            wait_async();
            __syncthreads();

            for (int ksi = 0; ksi < NSL; ++ksi) {
                const int ks  = ksi * KSL;
                const int buf = ksi & 1;

                // issue async DMA of next slice into the other buffer
                if (ksi + 1 < NSL) {
                    const v8bf* src =
                        (const v8bf*)(stageL + (size_t)((ks + KSL) >> 3) * BB * 8);
                    const unsigned dst = panel_base + (buf ^ 1) * SLICE_BYTES;
#pragma unroll
                    for (int i = 0; i < SLICE_VECS / NTHR; ++i) {
                        const unsigned vi = tid + i * NTHR;
                        async_ld_b128(dst + vi * 16, src + vi);
                    }
                }

                // compute current slice (DMA of next slice in flight)
                for (int k0 = ks; k0 < ks + KSL; k0 += 32) {
                    if (k0 + 32 < K2) {
                        __builtin_prefetch(wrow0 + k0 + 32, 0, 3);
                        __builtin_prefetch(wrow1 + k0 + 32, 0, 3);
                    }
                    // A fragments (16x32 bf16): lanes 0-15 K +0..7/+16..23,
                    //                           lanes 16-31 K +8..15/+24..31
                    const v8bf a00 = *(const v8bf*)(wrow0 + k0 + hi * 8);
                    const v8bf a01 = *(const v8bf*)(wrow0 + k0 + 16 + hi * 8);
                    const v16bf a0 = __builtin_shufflevector(
                        a00, a01, 0,1,2,3,4,5,6,7,8,9,10,11,12,13,14,15);
                    const v8bf a10 = *(const v8bf*)(wrow1 + k0 + hi * 8);
                    const v8bf a11 = *(const v8bf*)(wrow1 + k0 + 16 + hi * 8);
                    const v16bf a1 = __builtin_shufflevector(
                        a10, a11, 0,1,2,3,4,5,6,7,8,9,10,11,12,13,14,15);

                    const int g = (k0 - ks + hi * 8) >> 3;  // LDS K-group
                    const v8bf* bp0 = (const v8bf*)&panel[buf][g][ch * 32 + colb][0];
                    const v8bf* bp1 = (const v8bf*)&panel[buf][g][ch * 32 + 16 + colb][0];
                    const v16bf b0 = __builtin_shufflevector(
                        bp0[0], bp0[2 * BB], 0,1,2,3,4,5,6,7,8,9,10,11,12,13,14,15);
                    const v16bf b1 = __builtin_shufflevector(
                        bp1[0], bp1[2 * BB], 0,1,2,3,4,5,6,7,8,9,10,11,12,13,14,15);

                    acc00 = __builtin_amdgcn_wmma_f32_16x16x32_bf16(
                        false, a0, false, b0, (short)0, acc00, false, false);
                    acc01 = __builtin_amdgcn_wmma_f32_16x16x32_bf16(
                        false, a0, false, b1, (short)0, acc01, false, false);
                    acc10 = __builtin_amdgcn_wmma_f32_16x16x32_bf16(
                        false, a1, false, b0, (short)0, acc10, false, false);
                    acc11 = __builtin_amdgcn_wmma_f32_16x16x32_bf16(
                        false, a1, false, b1, (short)0, acc11, false, false);
                }

                if (ksi + 1 < NSL) {
                    wait_async();      // next slice landed in LDS
                    __syncthreads();   // ... for every wave's lanes
                }
            }

            // spill D tiles to LDS: row = d + 8*hi within each 16-row tile
#pragma unroll
            for (int d = 0; d < 8; ++d) {
                const int r0 = 0 * 16 + hi * 8 + d;
                const int r1 = 1 * 16 + hi * 8 + d;
                gl[s][r0][ch * 32 + colb]      = acc00[d];
                gl[s][r0][ch * 32 + 16 + colb] = acc01[d];
                gl[s][r1][ch * 32 + colb]      = acc10[d];
                gl[s][r1][ch * 32 + 16 + colb] = acc11[d];
            }
            grid_sync(bar, epoch);   // all blocks done reading stage[l]

            // ---- phase C: fused LSTM cell update (fp32) ----
            {
                const int ml = tid >> 3;            // 0..31 local m
                const int b0i = (tid & 7) * 8;      // 8 batch entries each
                const int m = bid * 32 + ml;
                const float bi = bias[l * G4 + 0 * MM + m];
                const float bf = bias[l * G4 + 1 * MM + m];
                const float bg = bias[l * G4 + 2 * MM + m];
                const float bo = bias[l * G4 + 3 * MM + m];
#pragma unroll
                for (int j = 0; j < 8; ++j) {
                    const int b = b0i + j;
                    const float si = sigm(gl[0][ml][b] + bi);
                    const float sf = sigm(gl[1][ml][b] + bf);
                    const float tg = tanh_fast(gl[2][ml][b] + bg);
                    const float so = sigm(gl[3][ml][b] + bo);
                    const size_t cidx = ((size_t)l * BB + b) * MM + m;
                    const float c_new = sf * c_ws[cidx] + si * tg;
                    c_ws[cidx] = c_new;
                    const float h_new = so * tanh_fast(c_new);
                    const __bf16 hb = (__bf16)h_new;
                    if (l == 0) {
                        stage[stg(1, m, b)]      = hb;  // input to layer 1
                        stage[stg(0, MM + m, b)] = hb;  // h_prev for t+1
                    } else {
                        stage[stg(1, MM + m, b)] = hb;  // h_prev for t+1
                        out[((size_t)b * TT + t) * MM + m] = h_new;
                    }
                    if (t == TT - 1) {
                        out[OUTS + cidx]       = h_new; // h_n
                        out[OUTS + LBM + cidx] = c_new; // c_n
                    }
                }
            }
            grid_sync(bar, epoch);   // staged h visible before next GEMM
        }
    }
}

extern "C" void kernel_launch(void* const* d_in, const int* in_sizes, int n_in,
                              void* d_out, int out_size, void* d_ws, size_t ws_size,
                              hipStream_t stream) {
    const float* x    = (const float*)d_in[0];   // (B,T,N)
    const float* h0   = (const float*)d_in[1];   // (L,B,M)
    const float* c0   = (const float*)d_in[2];   // (L,B,M)
    const float* Wx   = (const float*)d_in[3];   // (L,4M,M)
    const float* Wh   = (const float*)d_in[4];   // (L,4M,M)
    const float* bias = (const float*)d_in[5];   // (L,4M)
    float* out = (float*)d_out;

    char* ws = (char*)d_ws;
    const size_t wcat_bytes  = (size_t)LL * G4 * K2 * sizeof(__bf16); // 33.5 MB
    const size_t stage_bytes = (size_t)LL * K2 * BB * sizeof(__bf16); // 512 KB
    const size_t cws_bytes   = (size_t)LL * BB * MM * sizeof(float);  // 512 KB
    __bf16*  Wcat  = (__bf16*)ws;
    __bf16*  stage = (__bf16*)(ws + wcat_bytes);
    float*   c_ws  = (float*)(ws + wcat_bytes + stage_bytes);
    unsigned* bar  = (unsigned*)(ws + wcat_bytes + stage_bytes + cws_bytes);

    init_bar<<<1, 1, 0, stream>>>(bar);
    convert_weights<<<4096, NTHR, 0, stream>>>(Wx, Wh, Wcat);
    lstm_persistent<<<NBLK, NTHR, 0, stream>>>(x, h0, c0, bias, Wcat, stage,
                                               c_ws, bar, out);
}
